// Fractal_EGNN_18279380812419
// MI455X (gfx1250) — compile-verified, matrix operand order checked by reading, not run on hardware
//
#include <hip/hip_runtime.h>
#include <hip/hip_bf16.h>

#define NN   20000
#define EE   100000
#define DD   128
#define GG   64
#define LL   12
#define EPSV 1e-5f
#define LSTR 136   // padded LDS row stride in halfs (128+8) to avoid bank conflicts

typedef __attribute__((ext_vector_type(16))) _Float16 v16h;
typedef __attribute__((ext_vector_type(8)))  _Float16 v8h;
typedef __attribute__((ext_vector_type(8)))  float    v8f;

// ---------------------------------------------------------------------------
// WMMA wrapper: D = A(16x32 f16) * B(32x16 f16) + C(16x16 f32)
// ---------------------------------------------------------------------------
__device__ __forceinline__ v8f wmma16(v16h a, v16h b, v8f c) {
  return __builtin_amdgcn_wmma_f32_16x16x32_f16(false, a, false, b, (short)0, c,
                                                false, false);
}

// A-fragment from an fp32 row in global memory (row = lane&15, hh = lane>>4).
// 16-bit A 16x32 layout: lanes 0-15 hold K in {hh*8+0..7, 16+hh*8+0..7}.
__device__ __forceinline__ v16h a_frag_f32(const float* __restrict__ rowp, int k0, int hh) {
  const float4* p0 = (const float4*)(rowp + k0 + hh * 8);
  const float4* p1 = (const float4*)(rowp + k0 + 16 + hh * 8);
  float4 a0 = p0[0], a1 = p0[1], b0 = p1[0], b1 = p1[1];
  v16h r;
  r[0]  = (_Float16)a0.x; r[1]  = (_Float16)a0.y; r[2]  = (_Float16)a0.z; r[3]  = (_Float16)a0.w;
  r[4]  = (_Float16)a1.x; r[5]  = (_Float16)a1.y; r[6]  = (_Float16)a1.z; r[7]  = (_Float16)a1.w;
  r[8]  = (_Float16)b0.x; r[9]  = (_Float16)b0.y; r[10] = (_Float16)b0.z; r[11] = (_Float16)b0.w;
  r[12] = (_Float16)b1.x; r[13] = (_Float16)b1.y; r[14] = (_Float16)b1.z; r[15] = (_Float16)b1.w;
  return r;
}

// A-fragment from f16 tile staged in LDS (two 16B ds loads).
__device__ __forceinline__ v16h a_frag_lds(const _Float16* base, int row, int k0, int hh) {
  v8h lo = *(const v8h*)(base + row * LSTR + k0 + hh * 8);
  v8h hi = *(const v8h*)(base + row * LSTR + k0 + 16 + hh * 8);
  return __builtin_shufflevector(lo, hi, 0, 1, 2, 3, 4, 5, 6, 7, 8, 9, 10, 11, 12, 13, 14, 15);
}

// LayerNorm(+affine)+SiLU applied in-place to a 16x128 tile held in C-layout
// accumulators: acc[nt][v] = value at row M=v+8*hh, col n=nt*16+ln.
__device__ __forceinline__ void ln_silu(v8f acc[8], const float* __restrict__ g,
                                        const float* __restrict__ b, int ln) {
  float s[8], s2[8];
#pragma unroll
  for (int v = 0; v < 8; v++) { s[v] = 0.f; s2[v] = 0.f; }
#pragma unroll
  for (int nt = 0; nt < 8; nt++)
#pragma unroll
    for (int v = 0; v < 8; v++) { float x = acc[nt][v]; s[v] += x; s2[v] += x * x; }
#pragma unroll
  for (int m = 1; m < 16; m <<= 1)
#pragma unroll
    for (int v = 0; v < 8; v++) {
      s[v]  += __shfl_xor(s[v],  m, 16);
      s2[v] += __shfl_xor(s2[v], m, 16);
    }
  float mean[8], rs[8];
#pragma unroll
  for (int v = 0; v < 8; v++) {
    mean[v]   = s[v] * (1.f / 128.f);
    float var = s2[v] * (1.f / 128.f) - mean[v] * mean[v];
    rs[v]     = rsqrtf(var + EPSV);
  }
#pragma unroll
  for (int nt = 0; nt < 8; nt++) {
    float gg = g[nt * 16 + ln], bb = b[nt * 16 + ln];
#pragma unroll
    for (int v = 0; v < 8; v++) {
      float x = (acc[nt][v] - mean[v]) * rs[v] * gg + bb;
      acc[nt][v] = x / (1.f + __expf(-x));  // SiLU
    }
  }
}

// ---------------------------------------------------------------------------
// Weight pre-pack: fp32 [L, rowsPerLayer, 128] -> f16 B-fragment-swizzled.
// Tile (kt,nt) is 32x16; lane's 16 halfs stored contiguously (one 32B load).
// ---------------------------------------------------------------------------
__global__ void pack_kernel(const float* __restrict__ W, _Float16* __restrict__ pk,
                            int KT, int rowsPerLayer) {
  int i = blockIdx.x * blockDim.x + threadIdx.x;
  int total = LL * KT * 4096;  // KT*8 tiles * 512 halfs per layer
  if (i >= total) return;
  int j     = i & 15;
  int lane  = (i >> 4) & 31;
  int nt    = (i >> 9) & 7;
  int rest  = i >> 12;
  int kt    = rest % KT;
  int layer = rest / KT;
  int hh = lane >> 4, lnn = lane & 15;
  int kin = (j < 8) ? (hh * 8 + j) : (16 + hh * 8 + (j - 8));
  int k = kt * 32 + kin;
  int n = nt * 16 + lnn;
  pk[i] = (_Float16)W[((size_t)layer * rowsPerLayer + k) * DD + n];
}

// ---------------------------------------------------------------------------
// h = x @ emb_W + emb_b   (N x 16) * (16 x 128)
// ---------------------------------------------------------------------------
__global__ void embed_kernel(const float* __restrict__ x, const float* __restrict__ W,
                             const float* __restrict__ b, float* __restrict__ h) {
  int i = blockIdx.x * blockDim.x + threadIdx.x;
  if (i >= NN * DD) return;
  int node = i >> 7, d = i & 127;
  float s = b[d];
#pragma unroll
  for (int k = 0; k < 16; k++) s += x[node * 16 + k] * W[k * DD + d];
  h[i] = s;
}

// ---------------------------------------------------------------------------
// Fused edge-message kernel: one wave per 16-edge tile.
//   m = silu(ln1(cat(h[dst],h[src],d) @ W1 + b1))
//   m = silu(ln2(m @ W2 + b2));   agg[dst] += m  (fp32 atomics)
// ---------------------------------------------------------------------------
__global__ void __launch_bounds__(32)
msg_kernel(const float* __restrict__ h, const int* __restrict__ eidx,
           const float* __restrict__ pos,
           const _Float16* __restrict__ pkW1, const float* __restrict__ wlast,
           const float* __restrict__ b1, const float* __restrict__ g1,
           const float* __restrict__ be1,
           const _Float16* __restrict__ pkW2,
           const float* __restrict__ b2, const float* __restrict__ g2,
           const float* __restrict__ be2,
           float* __restrict__ agg) {
  __shared__ __align__(16) _Float16 sM[16 * LSTR];
  __shared__ float sDist[16];
  __shared__ int   sDst[16];

  const int lane = threadIdx.x;
  const int ln = lane & 15, hh = lane >> 4;
  const int e  = blockIdx.x * 16 + ln;
  const int si = eidx[e];        // row 0: src
  const int di = eidx[EE + e];   // row 1: dst

  float dx = pos[di * 3 + 0] - pos[si * 3 + 0];
  float dy = pos[di * 3 + 1] - pos[si * 3 + 1];
  float dz = pos[di * 3 + 2] - pos[si * 3 + 2];
  if (hh == 0) { sDist[ln] = sqrtf(dx * dx + dy * dy + dz * dz); sDst[ln] = di; }
  __syncthreads();

  v8f acc[8];
#pragma unroll
  for (int nt = 0; nt < 8; nt++)
#pragma unroll
    for (int v = 0; v < 8; v++) acc[nt][v] = 0.f;

  const float* rowd = h + (size_t)di * DD;
  const float* rows = h + (size_t)si * DD;
  // GEMM1 part A: h[dst] @ W1[0:128]
#pragma unroll
  for (int ks = 0; ks < 4; ks++) {
    v16h A = a_frag_f32(rowd, ks * 32, hh);
#pragma unroll
    for (int nt = 0; nt < 8; nt++) {
      v16h B = *(const v16h*)(pkW1 + ((ks * 8 + nt) << 9) + (lane << 4));
      acc[nt] = wmma16(A, B, acc[nt]);
    }
  }
  // GEMM1 part B: h[src] @ W1[128:256]
#pragma unroll
  for (int ks = 0; ks < 4; ks++) {
    v16h A = a_frag_f32(rows, ks * 32, hh);
#pragma unroll
    for (int nt = 0; nt < 8; nt++) {
      v16h B = *(const v16h*)(pkW1 + (((4 + ks) * 8 + nt) << 9) + (lane << 4));
      acc[nt] = wmma16(A, B, acc[nt]);
    }
  }
  // rank-1 distance term (W1 row 256, fp32) + bias
#pragma unroll
  for (int nt = 0; nt < 8; nt++) {
    float wl = wlast[nt * 16 + ln];
    float bb = b1[nt * 16 + ln];
#pragma unroll
    for (int v = 0; v < 8; v++) acc[nt][v] += sDist[v + 8 * hh] * wl + bb;
  }
  ln_silu(acc, g1, be1, ln);

  // stage m1 as f16 in LDS (C-layout -> row-major) for GEMM2 A-fragments
#pragma unroll
  for (int nt = 0; nt < 8; nt++)
#pragma unroll
    for (int v = 0; v < 8; v++)
      sM[(v + 8 * hh) * LSTR + nt * 16 + ln] = (_Float16)acc[nt][v];
  __syncthreads();

  v8f acc2[8];
#pragma unroll
  for (int nt = 0; nt < 8; nt++)
#pragma unroll
    for (int v = 0; v < 8; v++) acc2[nt][v] = 0.f;
#pragma unroll
  for (int ks = 0; ks < 4; ks++) {
    v16h A = a_frag_lds(sM, ln, ks * 32, hh);
#pragma unroll
    for (int nt = 0; nt < 8; nt++) {
      v16h B = *(const v16h*)(pkW2 + ((ks * 8 + nt) << 9) + (lane << 4));
      acc2[nt] = wmma16(A, B, acc2[nt]);
    }
  }
#pragma unroll
  for (int nt = 0; nt < 8; nt++) {
    float bb = b2[nt * 16 + ln];
#pragma unroll
    for (int v = 0; v < 8; v++) acc2[nt][v] += bb;
  }
  ln_silu(acc2, g2, be2, ln);

  // segment-sum scatter into agg[dst]
#pragma unroll
  for (int nt = 0; nt < 8; nt++)
#pragma unroll
    for (int v = 0; v < 8; v++)
      atomicAdd(&agg[(size_t)sDst[v + 8 * hh] * DD + nt * 16 + ln], acc2[nt][v]);
}

// ---------------------------------------------------------------------------
// Fused node-update kernel: one wave per 16-node tile.
//   u = silu(ln1(cat(h,agg) @ U1 + b1)); u = silu(ln2(u @ U2 + b2)); h += u
// ---------------------------------------------------------------------------
__global__ void __launch_bounds__(32)
upd_kernel(float* __restrict__ h, const float* __restrict__ agg,
           const _Float16* __restrict__ pkU1,
           const float* __restrict__ b1, const float* __restrict__ g1,
           const float* __restrict__ be1,
           const _Float16* __restrict__ pkU2,
           const float* __restrict__ b2, const float* __restrict__ g2,
           const float* __restrict__ be2) {
  __shared__ __align__(16) _Float16 sM[16 * LSTR];
  const int lane = threadIdx.x;
  const int ln = lane & 15, hh = lane >> 4;
  const int row = blockIdx.x * 16 + ln;
  const float* rh = h + (size_t)row * DD;
  const float* ra = agg + (size_t)row * DD;

  v8f acc[8];
#pragma unroll
  for (int nt = 0; nt < 8; nt++)
#pragma unroll
    for (int v = 0; v < 8; v++) acc[nt][v] = 0.f;

#pragma unroll
  for (int ks = 0; ks < 4; ks++) {
    v16h A = a_frag_f32(rh, ks * 32, hh);
#pragma unroll
    for (int nt = 0; nt < 8; nt++) {
      v16h B = *(const v16h*)(pkU1 + ((ks * 8 + nt) << 9) + (lane << 4));
      acc[nt] = wmma16(A, B, acc[nt]);
    }
  }
#pragma unroll
  for (int ks = 0; ks < 4; ks++) {
    v16h A = a_frag_f32(ra, ks * 32, hh);
#pragma unroll
    for (int nt = 0; nt < 8; nt++) {
      v16h B = *(const v16h*)(pkU1 + (((4 + ks) * 8 + nt) << 9) + (lane << 4));
      acc[nt] = wmma16(A, B, acc[nt]);
    }
  }
#pragma unroll
  for (int nt = 0; nt < 8; nt++) {
    float bb = b1[nt * 16 + ln];
#pragma unroll
    for (int v = 0; v < 8; v++) acc[nt][v] += bb;
  }
  ln_silu(acc, g1, be1, ln);

#pragma unroll
  for (int nt = 0; nt < 8; nt++)
#pragma unroll
    for (int v = 0; v < 8; v++)
      sM[(v + 8 * hh) * LSTR + nt * 16 + ln] = (_Float16)acc[nt][v];
  __syncthreads();

  v8f acc2[8];
#pragma unroll
  for (int nt = 0; nt < 8; nt++)
#pragma unroll
    for (int v = 0; v < 8; v++) acc2[nt][v] = 0.f;
#pragma unroll
  for (int ks = 0; ks < 4; ks++) {
    v16h A = a_frag_lds(sM, ln, ks * 32, hh);
#pragma unroll
    for (int nt = 0; nt < 8; nt++) {
      v16h B = *(const v16h*)(pkU2 + ((ks * 8 + nt) << 9) + (lane << 4));
      acc2[nt] = wmma16(A, B, acc2[nt]);
    }
  }
#pragma unroll
  for (int nt = 0; nt < 8; nt++) {
    float bb = b2[nt * 16 + ln];
#pragma unroll
    for (int v = 0; v < 8; v++) acc2[nt][v] += bb;
  }
  ln_silu(acc2, g2, be2, ln);

  // residual: this block exclusively owns rows [blockIdx.x*16, +16)
#pragma unroll
  for (int nt = 0; nt < 8; nt++)
#pragma unroll
    for (int v = 0; v < 8; v++) {
      size_t off = (size_t)(blockIdx.x * 16 + v + 8 * hh) * DD + nt * 16 + ln;
      h[off] += acc2[nt][v];
    }
}

// ---------------------------------------------------------------------------
// global_add_pool via atomics, then 2-layer prediction head.
// ---------------------------------------------------------------------------
__global__ void pool_kernel(const float* __restrict__ h, const int* __restrict__ batch,
                            float* __restrict__ pooled) {
  int i = blockIdx.x * blockDim.x + threadIdx.x;
  if (i >= NN * DD) return;
  int node = i >> 7, d = i & 127;
  atomicAdd(&pooled[(size_t)batch[node] * DD + d], h[i]);
}

__global__ void __launch_bounds__(128)
head_kernel(const float* __restrict__ pooled, const float* __restrict__ W1,
            const float* __restrict__ b1, const float* __restrict__ W2,
            const float* __restrict__ b2, float* __restrict__ out) {
  __shared__ float sp[128];
  __shared__ float st[128];
  int g = blockIdx.x, d = threadIdx.x;
  sp[d] = pooled[(size_t)g * DD + d];
  __syncthreads();
  float s = b1[d];
#pragma unroll 8
  for (int k = 0; k < 128; k++) s += sp[k] * W1[k * DD + d];
  st[d] = fmaxf(s, 0.f) * W2[d];
  __syncthreads();
  for (int off = 64; off > 0; off >>= 1) {
    if (d < off) st[d] += st[d + off];
    __syncthreads();
  }
  if (d == 0) out[g] = st[0] + b2[0];
}

// ---------------------------------------------------------------------------
extern "C" void kernel_launch(void* const* d_in, const int* in_sizes, int n_in,
                              void* d_out, int out_size, void* d_ws, size_t ws_size,
                              hipStream_t stream) {
  (void)in_sizes; (void)n_in; (void)out_size; (void)ws_size;
  const float* x     = (const float*)d_in[0];
  const float* pos   = (const float*)d_in[1];
  const int*   es[4] = {(const int*)d_in[2], (const int*)d_in[3],
                        (const int*)d_in[4], (const int*)d_in[5]};
  const int*   batch = (const int*)d_in[6];
  const float* embW  = (const float*)d_in[7];
  const float* embB  = (const float*)d_in[8];
  const float* msgW1 = (const float*)d_in[9];
  const float* msgB1 = (const float*)d_in[10];
  const float* msgG1 = (const float*)d_in[11];
  const float* msgE1 = (const float*)d_in[12];
  const float* msgW2 = (const float*)d_in[13];
  const float* msgB2 = (const float*)d_in[14];
  const float* msgG2 = (const float*)d_in[15];
  const float* msgE2 = (const float*)d_in[16];
  const float* updW1 = (const float*)d_in[17];
  const float* updB1 = (const float*)d_in[18];
  const float* updG1 = (const float*)d_in[19];
  const float* updE1 = (const float*)d_in[20];
  const float* updW2 = (const float*)d_in[21];
  const float* updB2 = (const float*)d_in[22];
  const float* updG2 = (const float*)d_in[23];
  const float* updE2 = (const float*)d_in[24];
  const float* prW1  = (const float*)d_in[25];
  const float* prB1  = (const float*)d_in[26];
  const float* prW2  = (const float*)d_in[27];
  const float* prB2  = (const float*)d_in[28];

  char* ws = (char*)d_ws;
  float* h      = (float*)ws; ws += (size_t)NN * DD * 4;
  float* agg    = (float*)ws; ws += (size_t)NN * DD * 4;
  float* pooled = (float*)ws; ws += (size_t)GG * DD * 4;
  _Float16* pkM1 = (_Float16*)ws; ws += (size_t)LL * 8 * 4096 * 2;
  _Float16* pkM2 = (_Float16*)ws; ws += (size_t)LL * 4 * 4096 * 2;
  _Float16* pkU1 = (_Float16*)ws; ws += (size_t)LL * 8 * 4096 * 2;
  _Float16* pkU2 = (_Float16*)ws; ws += (size_t)LL * 4 * 4096 * 2;

  // pre-swizzle weights to f16 B-fragment layout
  pack_kernel<<<(LL * 8 * 4096 + 255) / 256, 256, 0, stream>>>(msgW1, pkM1, 8, 257);
  pack_kernel<<<(LL * 4 * 4096 + 255) / 256, 256, 0, stream>>>(msgW2, pkM2, 4, 128);
  pack_kernel<<<(LL * 8 * 4096 + 255) / 256, 256, 0, stream>>>(updW1, pkU1, 8, 256);
  pack_kernel<<<(LL * 4 * 4096 + 255) / 256, 256, 0, stream>>>(updW2, pkU2, 4, 128);

  embed_kernel<<<(NN * DD + 255) / 256, 256, 0, stream>>>(x, embW, embB, h);

  for (int li = 0; li < LL; li++) {
    int gi = li & 3;
    hipMemsetAsync(agg, 0, (size_t)NN * DD * 4, stream);
    msg_kernel<<<EE / 16, 32, 0, stream>>>(
        h, es[gi], pos,
        pkM1 + (size_t)li * 8 * 4096,
        msgW1 + (size_t)li * 257 * DD + 256 * DD,  // W1 row 256 (distance coeffs)
        msgB1 + li * DD, msgG1 + li * DD, msgE1 + li * DD,
        pkM2 + (size_t)li * 4 * 4096,
        msgB2 + li * DD, msgG2 + li * DD, msgE2 + li * DD, agg);
    upd_kernel<<<NN / 16, 32, 0, stream>>>(
        h, agg,
        pkU1 + (size_t)li * 8 * 4096,
        updB1 + li * DD, updG1 + li * DD, updE1 + li * DD,
        pkU2 + (size_t)li * 4 * 4096,
        updB2 + li * DD, updG2 + li * DD, updE2 + li * DD);
  }

  hipMemsetAsync(pooled, 0, (size_t)GG * DD * 4, stream);
  pool_kernel<<<(NN * DD + 255) / 256, 256, 0, stream>>>(h, batch, pooled);
  head_kernel<<<GG, 128, 0, stream>>>(pooled, prW1, prB1, prW2, prB2, (float*)d_out);
}
